// MARNN_70815420776936
// MI455X (gfx1250) — matrix operand order repaired and checked by reading, not compiled, not run on hardware
//
#include <hip/hip_runtime.h>
#include <hip/hip_bf16.h>
#include <math.h>

// ---------------------------------------------------------------------------
// ARMiN scan for MI455X (gfx1250, wave32, WMMA).
// All four per-step GEMMs run on v_wmma_f32_16x16x32_bf16 with f32 accumulate.
// Each wave computes a 16x64 output strip (4 N-tiles) so the A fragment is
// reused across 4 WMMAs per K-chunk. Weights converted once to bf16, stored
// N-major (transposed) so B-fragments are contiguous 16-byte loads.
// Sequential scan handled by stream ordering (6 kernels/step, 256 steps).
// ---------------------------------------------------------------------------

typedef __bf16 bf16_t;
typedef __attribute__((ext_vector_type(16))) __bf16 v16bf;
typedef __attribute__((ext_vector_type(8)))  __bf16 v8bf;
typedef __attribute__((ext_vector_type(8)))  float  v8f;

#define XD 256
#define HD 512
#define RD 128
#define MD 128
#define BD 256
#define TD 256
#define CAT 896        // X + H + R
#define G1 640         // R + H (gates1 width)
#define PREW 2176      // R + 4H
#define OUTW 640       // H + R

__device__ __forceinline__ float sigf(float x) { return 1.0f / (1.0f + expf(-x)); }

// A fragment: 16x32 bf16 tile, rows m0..m0+15, cols k0..k0+31, row-major src.
// ISA layout: lane (l%16)=row, half=l/16; elems 0..7 = K[half*8 .. +8),
// elems 8..15 = K[16+half*8 .. +8).
__device__ __forceinline__ v16bf load_a_bf(const bf16_t* A, int lda, int m0, int k0, int lane) {
  int lm = lane & 15, lh = lane >> 4;
  const bf16_t* p = A + (size_t)(m0 + lm) * lda + k0 + lh * 8;
  v8bf lo = *reinterpret_cast<const v8bf*>(p);
  v8bf hi = *reinterpret_cast<const v8bf*>(p + 16);
  return __builtin_shufflevector(lo, hi, 0,1,2,3,4,5,6,7,8,9,10,11,12,13,14,15);
}

// Same A fragment but from an f32 source, converting to bf16 in-register.
__device__ __forceinline__ v16bf load_a_f32(const float* A, int lda, int m0, int k0, int lane) {
  int lm = lane & 15, lh = lane >> 4;
  const float* p = A + (size_t)(m0 + lm) * lda + k0 + lh * 8;
  v16bf r;
#pragma unroll
  for (int i = 0; i < 8; ++i) {
    r[i]     = (bf16_t)p[i];
    r[i + 8] = (bf16_t)p[i + 16];
  }
  return r;
}

// B fragment: 32x16 bf16 tile from an N-major (transposed, ldb=K) weight array.
// lane (l%16)=col n, half=l/16; elems 0..7 = K[half*8 .. +8), elems 8..15 = K[16+half*8 .. +8).
__device__ __forceinline__ v16bf load_b_bf(const bf16_t* BT, int ldb, int n0, int k0, int lane) {
  int lm = lane & 15, lh = lane >> 4;
  const bf16_t* p = BT + (size_t)(n0 + lm) * ldb + k0 + lh * 8;
  v8bf lo = *reinterpret_cast<const v8bf*>(p);
  v8bf hi = *reinterpret_cast<const v8bf*>(p + 16);
  return __builtin_shufflevector(lo, hi, 0,1,2,3,4,5,6,7,8,9,10,11,12,13,14,15);
}

__device__ __forceinline__ v8f wmma_bf16(v16bf a, v16bf b, v8f c) {
  return __builtin_amdgcn_wmma_f32_16x16x32_bf16(false, a, false, b, (short)0, c, false, false);
}

// ------------------------- preamble kernels --------------------------------

__global__ void k_cvt(const float* __restrict__ src, bf16_t* __restrict__ dst, int n) {
  int i = blockIdx.x * blockDim.x + threadIdx.x;
  if (i < n) dst[i] = (bf16_t)src[i];
}

// src (K,N) row-major f32 -> dst (N,K) row-major bf16
__global__ void k_cvtT(const float* __restrict__ src, bf16_t* __restrict__ dst, int K, int N) {
  int i = blockIdx.x * blockDim.x + threadIdx.x;
  if (i >= K * N) return;
  int k = i / N, n = i % N;
  dst[(size_t)n * K + k] = (bf16_t)src[i];
}

__global__ void k_init(const float* __restrict__ c_bias, const float* __restrict__ hmem_bias,
                       float* __restrict__ cf, bf16_t* __restrict__ cbf, float* __restrict__ hmem) {
  int i = blockIdx.x * blockDim.x + threadIdx.x;   // covers B*M*R = 4M elems
  if (i < BD * HD) {
    float v = tanhf(c_bias[i % HD]);
    cf[i] = v;
    cbf[i] = (bf16_t)v;
  }
  if (i < BD * MD * RD) hmem[i] = hmem_bias[i % (MD * RD)];
}

// ------------------------- per-step kernels --------------------------------

// (1a) head = [x_t | c] @ fc_w^T + fc_b + gumbel(noise).  [256x768]x[768x128]
// 16x64 strip per wave: 2 n-groups x 16 m-tiles = 32 waves.
__global__ __launch_bounds__(128) void k_head(const float* __restrict__ x_t,
                                              const bf16_t* __restrict__ cbf,
                                              const bf16_t* __restrict__ fcwT,
                                              const float* __restrict__ fc_b,
                                              const float* __restrict__ noise_t,
                                              float* __restrict__ head) {
  int wave = (blockIdx.x * blockDim.x + threadIdx.x) >> 5;
  int lane = threadIdx.x & 31;
  const int NG = MD / 64;                 // 2
  if (wave >= (BD / 16) * NG) return;
  int mt = wave / NG, n0 = (wave % NG) * 64;
  v8f acc0 = {}, acc1 = {}, acc2 = {}, acc3 = {};
  for (int k0 = 0; k0 < XD + HD; k0 += 32) {
    v16bf a = (k0 < XD) ? load_a_f32(x_t, XD, mt * 16, k0, lane)
                        : load_a_bf(cbf, HD, mt * 16, k0 - XD, lane);
    v16bf b0 = load_b_bf(fcwT, XD + HD, n0,      k0, lane);
    v16bf b1 = load_b_bf(fcwT, XD + HD, n0 + 16, k0, lane);
    v16bf b2 = load_b_bf(fcwT, XD + HD, n0 + 32, k0, lane);
    v16bf b3 = load_b_bf(fcwT, XD + HD, n0 + 48, k0, lane);
    acc0 = wmma_bf16(a, b0, acc0);
    acc1 = wmma_bf16(a, b1, acc1);
    acc2 = wmma_bf16(a, b2, acc2);
    acc3 = wmma_bf16(a, b3, acc3);
  }
  int lm = lane & 15, lh = lane >> 4;
  v8f accs[4] = {acc0, acc1, acc2, acc3};
#pragma unroll
  for (int tt = 0; tt < 4; ++tt) {
    int n = n0 + tt * 16 + lm;
    float bias = fc_b[n];
#pragma unroll
    for (int v = 0; v < 8; ++v) {
      int b_ = mt * 16 + lh * 8 + v;
      float g = noise_t[(size_t)b_ * MD + n];
      float gum = -logf(1e-20f - logf(1e-20f + g));
      head[(size_t)b_ * MD + n] = accs[tt][v] + bias + gum;
    }
  }
}

// (1b) per-row argmax -> idx; gather h_entry = hmem[b, idx]; assemble concat.
__global__ __launch_bounds__(128) void k_select(const float* __restrict__ head,
                                                const float* __restrict__ hmem,
                                                const float* __restrict__ x_t,
                                                const bf16_t* __restrict__ cbf,
                                                int* __restrict__ idxb,
                                                float* __restrict__ hent,
                                                bf16_t* __restrict__ concatb,
                                                bf16_t* __restrict__ gcatb) {
  int b = blockIdx.x;
  int tid = threadIdx.x;                  // 128 threads
  __shared__ float sv[128];
  __shared__ int si[128];
  sv[tid] = head[(size_t)b * MD + tid];
  si[tid] = tid;
  __syncthreads();
  for (int s = 64; s > 0; s >>= 1) {
    if (tid < s) {
      float v2 = sv[tid + s]; int i2 = si[tid + s];
      if (v2 > sv[tid] || (v2 == sv[tid] && i2 < si[tid])) { sv[tid] = v2; si[tid] = i2; }
    }
    __syncthreads();
  }
  int midx = si[0];
  if (tid == 0) idxb[b] = midx;
  // gather hmem row (128 floats)
  float hv = hmem[((size_t)b * MD + midx) * RD + tid];
  hent[(size_t)b * RD + tid] = hv;
  concatb[(size_t)b * CAT + XD + HD + tid] = (bf16_t)hv;
  // x part (256) -> concat and gcat (ungated)
  for (int j = tid; j < XD; j += 128) {
    bf16_t xv = (bf16_t)x_t[(size_t)b * XD + j];
    concatb[(size_t)b * CAT + j] = xv;
    gcatb[(size_t)b * CAT + j] = xv;
  }
  // c part (512)
  for (int j = tid; j < HD; j += 128)
    concatb[(size_t)b * CAT + XD + j] = cbf[(size_t)b * HD + j];
}

// (2) gates1 = sigmoid(concat @ W1 + b1); gcat[:,256:] = concat[:,256:]*gates1
// 16x64 strips: 10 n-groups x 16 m-tiles = 160 waves.
__global__ __launch_bounds__(128) void k_gates(const bf16_t* __restrict__ concatb,
                                               const bf16_t* __restrict__ W1T,
                                               const float* __restrict__ bias1,
                                               bf16_t* __restrict__ gcatb) {
  int wave = (blockIdx.x * blockDim.x + threadIdx.x) >> 5;
  int lane = threadIdx.x & 31;
  const int NG = G1 / 64;                 // 10
  if (wave >= (BD / 16) * NG) return;
  int mt = wave / NG, n0 = (wave % NG) * 64;
  v8f acc0 = {}, acc1 = {}, acc2 = {}, acc3 = {};
  for (int k0 = 0; k0 < CAT; k0 += 32) {
    v16bf a = load_a_bf(concatb, CAT, mt * 16, k0, lane);
    v16bf b0 = load_b_bf(W1T, CAT, n0,      k0, lane);
    v16bf b1 = load_b_bf(W1T, CAT, n0 + 16, k0, lane);
    v16bf b2 = load_b_bf(W1T, CAT, n0 + 32, k0, lane);
    v16bf b3 = load_b_bf(W1T, CAT, n0 + 48, k0, lane);
    acc0 = wmma_bf16(a, b0, acc0);
    acc1 = wmma_bf16(a, b1, acc1);
    acc2 = wmma_bf16(a, b2, acc2);
    acc3 = wmma_bf16(a, b3, acc3);
  }
  int lm = lane & 15, lh = lane >> 4;
  v8f accs[4] = {acc0, acc1, acc2, acc3};
#pragma unroll
  for (int tt = 0; tt < 4; ++tt) {
    int n = n0 + tt * 16 + lm;
    float bb = bias1[n];
#pragma unroll
    for (int v = 0; v < 8; ++v) {
      int b_ = mt * 16 + lh * 8 + v;
      float g = sigf(accs[tt][v] + bb);
      float cv = (float)concatb[(size_t)b_ * CAT + XD + n];
      gcatb[(size_t)b_ * CAT + XD + n] = (bf16_t)(cv * g);
    }
  }
}

// (3a) pre = gcat @ W_full + bias.  [256x896]x[896x2176]
// 16x64 strips: 34 n-groups x 16 m-tiles = 544 waves.
__global__ __launch_bounds__(128) void k_pre(const bf16_t* __restrict__ gcatb,
                                             const bf16_t* __restrict__ WT,
                                             const float* __restrict__ bias,
                                             float* __restrict__ pre) {
  int wave = (blockIdx.x * blockDim.x + threadIdx.x) >> 5;
  int lane = threadIdx.x & 31;
  const int NG = PREW / 64;               // 34
  if (wave >= (BD / 16) * NG) return;
  int mt = wave / NG, n0 = (wave % NG) * 64;
  v8f acc0 = {}, acc1 = {}, acc2 = {}, acc3 = {};
  for (int k0 = 0; k0 < CAT; k0 += 32) {
    v16bf a = load_a_bf(gcatb, CAT, mt * 16, k0, lane);
    v16bf b0 = load_b_bf(WT, CAT, n0,      k0, lane);
    v16bf b1 = load_b_bf(WT, CAT, n0 + 16, k0, lane);
    v16bf b2 = load_b_bf(WT, CAT, n0 + 32, k0, lane);
    v16bf b3 = load_b_bf(WT, CAT, n0 + 48, k0, lane);
    acc0 = wmma_bf16(a, b0, acc0);
    acc1 = wmma_bf16(a, b1, acc1);
    acc2 = wmma_bf16(a, b2, acc2);
    acc3 = wmma_bf16(a, b3, acc3);
  }
  int lm = lane & 15, lh = lane >> 4;
  v8f accs[4] = {acc0, acc1, acc2, acc3};
#pragma unroll
  for (int tt = 0; tt < 4; ++tt) {
    int n = n0 + tt * 16 + lm;
    float bb = bias[n];
#pragma unroll
    for (int v = 0; v < 8; ++v) {
      int b_ = mt * 16 + lh * 8 + v;
      pre[(size_t)b_ * PREW + n] = accs[tt][v] + bb;
    }
  }
}

// (3b) cell nonlinearity + outputs.
__global__ __launch_bounds__(256) void k_cell(const float* __restrict__ pre,
                                              const float* __restrict__ hent,
                                              float* __restrict__ cf,
                                              bf16_t* __restrict__ cbf,
                                              float* __restrict__ out_t) {
  int i = blockIdx.x * blockDim.x + threadIdx.x;
  if (i >= BD * OUTW) return;
  int b = i / OUTW, col = i % OUTW;
  const float* prow = pre + (size_t)b * PREW;
  if (col < HD) {
    float iv = prow[col];
    float jv = prow[HD + col];
    float fv = prow[2 * HD + col];
    float ov = prow[3 * HD + col];
    float c_old = cf[(size_t)b * HD + col];
    float nc = tanhf(c_old * sigf(fv + 1.0f) + sigf(iv) * tanhf(jv));
    cf[(size_t)b * HD + col] = nc;
    cbf[(size_t)b * HD + col] = (bf16_t)nc;
    out_t[(size_t)b * OUTW + col] = nc * sigf(ov);
  } else {
    int j = col - HD;
    float om = prow[4 * HD + j];
    out_t[(size_t)b * OUTW + col] = hent[(size_t)b * RD + j] * sigf(om);
  }
}

// (4) wv = new_c @ trans_w^T + trans_b; scatter into hmem[b, widx(b)].
// 16x64 strips: 2 n-groups x 16 m-tiles = 32 waves.
__global__ __launch_bounds__(128) void k_wv(const bf16_t* __restrict__ cbf,
                                            const bf16_t* __restrict__ transT,
                                            const float* __restrict__ trans_b,
                                            const int* __restrict__ idxb,
                                            float* __restrict__ hmem, int t) {
  int wave = (blockIdx.x * blockDim.x + threadIdx.x) >> 5;
  int lane = threadIdx.x & 31;
  const int NG = RD / 64;                 // 2
  if (wave >= (BD / 16) * NG) return;
  int mt = wave / NG, n0 = (wave % NG) * 64;
  v8f acc0 = {}, acc1 = {}, acc2 = {}, acc3 = {};
  for (int k0 = 0; k0 < HD; k0 += 32) {
    v16bf a = load_a_bf(cbf, HD, mt * 16, k0, lane);
    v16bf b0 = load_b_bf(transT, HD, n0,      k0, lane);
    v16bf b1 = load_b_bf(transT, HD, n0 + 16, k0, lane);
    v16bf b2 = load_b_bf(transT, HD, n0 + 32, k0, lane);
    v16bf b3 = load_b_bf(transT, HD, n0 + 48, k0, lane);
    acc0 = wmma_bf16(a, b0, acc0);
    acc1 = wmma_bf16(a, b1, acc1);
    acc2 = wmma_bf16(a, b2, acc2);
    acc3 = wmma_bf16(a, b3, acc3);
  }
  int lm = lane & 15, lh = lane >> 4;
  v8f accs[4] = {acc0, acc1, acc2, acc3};
#pragma unroll
  for (int tt = 0; tt < 4; ++tt) {
    int n = n0 + tt * 16 + lm;
    float bb = trans_b[n];
#pragma unroll
    for (int v = 0; v < 8; ++v) {
      int b_ = mt * 16 + lh * 8 + v;
      int widx = (t < MD) ? t : idxb[b_];
      hmem[((size_t)b_ * MD + widx) * RD + n] = accs[tt][v] + bb;
    }
  }
}

// ---------------------------------------------------------------------------

extern "C" void kernel_launch(void* const* d_in, const int* in_sizes, int n_in,
                              void* d_out, int out_size, void* d_ws, size_t ws_size,
                              hipStream_t stream) {
  const float* x         = (const float*)d_in[0];   // (T,B,X)
  const float* noise     = (const float*)d_in[1];   // (T,B,M)
  const float* W_full    = (const float*)d_in[2];   // (896,2176)
  const float* bias      = (const float*)d_in[3];   // (2176)
  const float* W_full1   = (const float*)d_in[4];   // (896,640)
  const float* bias1     = (const float*)d_in[5];   // (640)
  const float* fc_w      = (const float*)d_in[6];   // (128,768)  (already N-major for B^T)
  const float* fc_b      = (const float*)d_in[7];   // (128)
  const float* trans_w   = (const float*)d_in[8];   // (128,512)  (already N-major for B^T)
  const float* trans_b   = (const float*)d_in[9];   // (128)
  const float* c_bias    = (const float*)d_in[10];  // (1,512)
  const float* hmem_bias = (const float*)d_in[11];  // (1,128,128)
  float* out = (float*)d_out;

  // workspace carve-up (256B aligned)
  size_t off = 0;
  auto alloc = [&](size_t bytes) -> void* {
    void* p = (char*)d_ws + off;
    off += (bytes + 255) & ~(size_t)255;
    return p;
  };
  bf16_t* WT     = (bf16_t*)alloc((size_t)PREW * CAT * 2);  // W_full^T    3.9 MB
  bf16_t* W1T    = (bf16_t*)alloc((size_t)G1 * CAT * 2);    // W_full1^T   1.1 MB
  bf16_t* fcwT   = (bf16_t*)alloc((size_t)MD * (XD + HD) * 2);
  bf16_t* transT = (bf16_t*)alloc((size_t)RD * HD * 2);
  float*  cf     = (float*)alloc((size_t)BD * HD * 4);
  bf16_t* cbf    = (bf16_t*)alloc((size_t)BD * HD * 2);
  float*  hmem   = (float*)alloc((size_t)BD * MD * RD * 4); // 16 MB
  float*  head   = (float*)alloc((size_t)BD * MD * 4);
  int*    idxb   = (int*)alloc((size_t)BD * 4);
  float*  hent   = (float*)alloc((size_t)BD * RD * 4);
  bf16_t* catb   = (bf16_t*)alloc((size_t)BD * CAT * 2);
  bf16_t* gcatb  = (bf16_t*)alloc((size_t)BD * CAT * 2);
  float*  pre    = (float*)alloc((size_t)BD * PREW * 4);    // 2.2 MB
  (void)ws_size; (void)n_in; (void)in_sizes; (void)out_size;

  // ---- preamble: weight conversion + state init ----
  {
    int n1 = MD * (XD + HD);
    k_cvt<<<(n1 + 255) / 256, 256, 0, stream>>>(fc_w, fcwT, n1);
    int n2 = RD * HD;
    k_cvt<<<(n2 + 255) / 256, 256, 0, stream>>>(trans_w, transT, n2);
    int n3 = CAT * G1;
    k_cvtT<<<(n3 + 255) / 256, 256, 0, stream>>>(W_full1, W1T, CAT, G1);
    int n4 = CAT * PREW;
    k_cvtT<<<(n4 + 255) / 256, 256, 0, stream>>>(W_full, WT, CAT, PREW);
    int n5 = BD * MD * RD;
    k_init<<<(n5 + 255) / 256, 256, 0, stream>>>(c_bias, hmem_bias, cf, cbf, hmem);
  }

  // ---- sequential scan: 6 kernels per step, dependencies via stream order ----
  for (int t = 0; t < TD; ++t) {
    const float* x_t = x + (size_t)t * BD * XD;
    const float* n_t = noise + (size_t)t * BD * MD;
    float* out_t = out + (size_t)t * BD * OUTW;

    k_head  <<<(BD / 16) * (MD / 64) / 4, 128, 0, stream>>>(x_t, cbf, fcwT, fc_b, n_t, head);
    k_select<<<BD, 128, 0, stream>>>(head, hmem, x_t, cbf, idxb, hent, catb, gcatb);
    k_gates <<<(BD / 16) * (G1 / 64) / 4, 128, 0, stream>>>(catb, W1T, bias1, gcatb);
    k_pre   <<<(BD / 16) * (PREW / 64) / 4, 128, 0, stream>>>(gcatb, WT, bias, pre);
    k_cell  <<<(BD * OUTW + 255) / 256, 256, 0, stream>>>(pre, hent, cf, cbf, out_t);
    k_wv    <<<(BD / 16) * (RD / 64) / 4, 128, 0, stream>>>(cbf, transT, trans_b, idxb, hmem, t);
  }
}
